// Attention_48309792145474
// MI455X (gfx1250) — matrix operand order repaired and checked by reading, not compiled, not run on hardware
//
#include <hip/hip_runtime.h>
#include <hip/hip_bf16.h>
#include <math.h>
#include <stdint.h>

// ---------------- problem constants ----------------
#define BB      2
#define TLEN    2048
#define DIM     4096
#define NHEADS  32
#define NKVH    8
#define HD      128
#define NREP    (NHEADS / NKVH)
#define MTOK    (BB * TLEN)          // 4096 tokens
#define HALF_HD (HD / 2)

typedef __bf16 bf;
typedef __attribute__((ext_vector_type(16))) __bf16 v16bf;
typedef __attribute__((ext_vector_type(8)))  float  v8f;

union FragU { v16bf v; uint4 q[2]; };

__device__ __forceinline__ v8f zero8() {
  v8f z = {0.f, 0.f, 0.f, 0.f, 0.f, 0.f, 0.f, 0.f};
  return z;
}

// A fragment: 16x32 bf16, row-major source with leading dim ld.
// Lane L: row m = L&15, half = L>>4.  Element i: k = (i/8)*16 + half*8 + (i%8).
__device__ __forceinline__ v16bf load_frag_a(const bf* base, int ld) {
  int lane = threadIdx.x & 31;
  int m = lane & 15, half = lane >> 4;
  const bf* row = base + (size_t)m * ld + half * 8;
  FragU f;
  f.q[0] = *(const uint4*)(row);        // k = half*8 + 0..7
  f.q[1] = *(const uint4*)(row + 16);   // k = 16 + half*8 + 0..7
  return f.v;
}

// B fragment: 32x16 bf16 sourced from [n][k] row-major (i.e. B^T) with ld.
// Lane L: col n = L&15, half = L>>4.  Element i: k = (i/8)*16 + half*8 + (i%8).
__device__ __forceinline__ v16bf load_frag_b(const bf* baseT, int ld) {
  int lane = threadIdx.x & 31;
  int n = lane & 15, half = lane >> 4;
  const bf* row = baseT + (size_t)n * ld + half * 8;
  FragU f;
  f.q[0] = *(const uint4*)(row);        // k = half*8 + 0..7
  f.q[1] = *(const uint4*)(row + 16);   // k = 16 + half*8 + 0..7
  return f.v;
}

__device__ __forceinline__ v8f wmma_bf16(v16bf a, v16bf b, v8f c) {
  return __builtin_amdgcn_wmma_f32_16x16x32_bf16(
      /*neg_a=*/false, a, /*neg_b=*/false, b,
      /*c_mod=*/(short)0, c, /*reuse_a=*/false, /*reuse_b=*/false);
}

// C/D fragment store: lane L -> n = L&15, vgpr j -> m = j + (L>>4)*8
__device__ __forceinline__ void store_c_f32(float* __restrict__ C, int ldc, v8f c) {
  int lane = threadIdx.x & 31;
  int n = lane & 15, half = lane >> 4;
#pragma unroll
  for (int j = 0; j < 8; ++j) C[(size_t)(j + half * 8) * ldc + n] = c[j];
}
__device__ __forceinline__ void store_c_bf16(bf* __restrict__ C, int ldc, v8f c) {
  int lane = threadIdx.x & 31;
  int n = lane & 15, half = lane >> 4;
#pragma unroll
  for (int j = 0; j < 8; ++j) C[(size_t)(j + half * 8) * ldc + n] = (bf)c[j];
}

// CDNA5 async copy: global -> LDS, 16 bytes per lane, tracked by ASYNCcnt.
// VDST VGPR holds the LDS byte offset (low 32 bits of the flat shared address,
// per the LDS aperture rule).
__device__ __forceinline__ void async_load_b128(uint32_t lds_off, const void* gptr) {
  asm volatile("global_load_async_to_lds_b128 %0, %1, off"
               :: "v"(lds_off), "v"((uint64_t)(uintptr_t)gptr)
               : "memory");
}
__device__ __forceinline__ void wait_async0() {
  asm volatile("s_wait_asynccnt 0" ::: "memory");
}

// ---------------- conversion kernels ----------------
__global__ void __launch_bounds__(256)
k_f32_to_bf16(const float* __restrict__ in, bf* __restrict__ out, size_t n) {
  size_t i = (size_t)blockIdx.x * blockDim.x + threadIdx.x;
  if (i < n) out[i] = (bf)in[i];
}

// in: [K][N] f32  ->  out: [N][K] bf16 (weight transpose so GEMM B is contiguous-K)
__global__ void __launch_bounds__(256)
k_transpose_to_bf16(const float* __restrict__ in, bf* __restrict__ out, int K, int N) {
  size_t i = (size_t)blockIdx.x * blockDim.x + threadIdx.x;
  size_t total = (size_t)K * N;
  if (i >= total) return;
  int k = (int)(i % K);
  int n = (int)(i / K);
  out[(size_t)n * K + k] = (bf)in[(size_t)k * N + n];
}

// ---------------- WMMA GEMM with async-LDS double buffering ----------------
// C[M,N] = A[M,K] * B(stored as [N][K]).
// 8 waves (256 thr); WG tile 128x128; stage depth 64 (two k=32 substeps).
// Async engine DMAs the next 32KB stage while the waves run 16 WMMAs each
// from the current LDS stage (ds_load_b128 fragments, full WG reuse).
template <bool STORE_F32>
__global__ void __launch_bounds__(256)
k_gemm_wmma(const bf* __restrict__ A, const bf* __restrict__ Bt,
            void* __restrict__ Cout, int M, int N, int K) {
  __shared__ bf Abuf[2][128 * 64];   // [row m][k], 16 KB per buffer
  __shared__ bf Bbuf[2][128 * 64];   // [row n][k], 16 KB per buffer

  const int tid  = threadIdx.x;
  const int wave = tid >> 5;
  const int m0 = blockIdx.y * 128;
  const int n0 = blockIdx.x * 128;
  const int wm = (wave >> 1) * 32;   // wave sub-tile origin: 32 rows
  const int wn = (wave & 1) * 64;    // 64 cols

  // Each thread DMAs 4x16B of A and 4x16B of B per stage
  // (128 rows x 8 chunks = 1024 chunks of 16B per matrix, 256 threads).
  auto issue_stage = [&](int k0, int buf) {
#pragma unroll
    for (int u = 0; u < 4; ++u) {
      int c = tid * 4 + u;
      int row = c >> 3, cc = c & 7;          // row 0..127, 8x 16B chunks per row
      async_load_b128((uint32_t)(uintptr_t)&Abuf[buf][row * 64 + cc * 8],
                      A + (size_t)(m0 + row) * K + k0 + cc * 8);
      async_load_b128((uint32_t)(uintptr_t)&Bbuf[buf][row * 64 + cc * 8],
                      Bt + (size_t)(n0 + row) * K + k0 + cc * 8);
    }
  };

  v8f acc[2][4];
#pragma unroll
  for (int i = 0; i < 2; ++i)
#pragma unroll
    for (int j = 0; j < 4; ++j) acc[i][j] = zero8();

  int cur = 0;
  issue_stage(0, 0);
  for (int k0 = 0; k0 < K; k0 += 64) {
    wait_async0();          // this wave's DMA for stage `cur` complete
    __syncthreads();        // ...and every wave's -> stage visible in LDS
    if (k0 + 64 < K) issue_stage(k0 + 64, cur ^ 1);  // overlap next stage's DMA

#pragma unroll
    for (int ks = 0; ks < 64; ks += 32) {
      v16bf a0 = load_frag_a(&Abuf[cur][(wm +  0) * 64 + ks], 64);
      v16bf a1 = load_frag_a(&Abuf[cur][(wm + 16) * 64 + ks], 64);
#pragma unroll
      for (int nt = 0; nt < 4; ++nt) {
        v16bf bfr = load_frag_b(&Bbuf[cur][(wn + nt * 16) * 64 + ks], 64);
        acc[0][nt] = wmma_bf16(a0, bfr, acc[0][nt]);
        acc[1][nt] = wmma_bf16(a1, bfr, acc[1][nt]);
      }
    }

    __syncthreads();        // all waves done reading stage `cur` before reuse
    cur ^= 1;
  }

  const int mo = m0 + wm, no = n0 + wn;
#pragma unroll
  for (int mi = 0; mi < 2; ++mi) {
#pragma unroll
    for (int nt = 0; nt < 4; ++nt) {
      if (STORE_F32) {
        float* C = (float*)Cout;
        store_c_f32(C + (size_t)(mo + mi * 16) * N + no + nt * 16, N, acc[mi][nt]);
      } else {
        bf* C = (bf*)Cout;
        store_c_bf16(C + (size_t)(mo + mi * 16) * N + no + nt * 16, N, acc[mi][nt]);
      }
    }
  }
}

// ---------------- RoPE + reshape to [B, heads, T, HD] (bf16) ----------------
__global__ void __launch_bounds__(256)
k_rope_reshape(const bf* __restrict__ lin, const float* __restrict__ freqs,
               bf* __restrict__ out, int nheads) {
  size_t idx = (size_t)blockIdx.x * blockDim.x + threadIdx.x;
  size_t total = (size_t)BB * TLEN * nheads * HALF_HD;
  if (idx >= total) return;
  int d = (int)(idx % HALF_HD);
  int h = (int)((idx / HALF_HD) % nheads);
  int t = (int)((idx / ((size_t)HALF_HD * nheads)) % TLEN);
  int b = (int)(idx / ((size_t)HALF_HD * nheads * TLEN));
  const bf* src = lin + ((size_t)(b * TLEN + t)) * (nheads * HD) + h * HD + 2 * d;
  float xr = (float)src[0];
  float xi = (float)src[1];
  float f = freqs[(size_t)t * HALF_HD + d];
  float c = __cosf(f), s = __sinf(f);
  bf* dst = out + (((size_t)(b * nheads + h) * TLEN + t)) * HD + 2 * d;
  dst[0] = (bf)(xr * c - xi * s);
  dst[1] = (bf)(xr * s + xi * c);
}

// ---------------- V transpose: [B*T, NKVH*HD] -> [B, NKVH, HD, T] ----------------
__global__ void __launch_bounds__(256)
k_v_transpose(const bf* __restrict__ lin, bf* __restrict__ out) {
  size_t idx = (size_t)blockIdx.x * blockDim.x + threadIdx.x;
  size_t total = (size_t)BB * TLEN * NKVH * HD;
  if (idx >= total) return;
  int d   = (int)(idx % HD);
  int kvh = (int)((idx / HD) % NKVH);
  int t   = (int)((idx / (HD * NKVH)) % TLEN);
  int b   = (int)(idx / ((size_t)HD * NKVH * TLEN));
  out[((size_t)(b * NKVH + kvh) * HD + d) * TLEN + t] =
      lin[((size_t)(b * TLEN + t)) * (NKVH * HD) + kvh * HD + d];
}

// ---------------- flash attention: 1 wave per 16-query block ----------------
// Q: [B,NHEADS,T,HD] bf16 ; K: [B,NKVH,T,HD] bf16 ; Vt: [B,NKVH,HD,T] bf16
// Out: [B*T, DIM] bf16 (token-major, ready for the Wo GEMM)
__global__ void __launch_bounds__(32)
k_flash_attn(const bf* __restrict__ Q, const bf* __restrict__ Kc,
             const bf* __restrict__ Vt, bf* __restrict__ Out) {
  __shared__ bf Pbuf[16 * 32];

  const int lane = threadIdx.x & 31;
  const int n = lane & 15, half = lane >> 4;
  const int qblocks = TLEN / 16;
  int qb = blockIdx.x % qblocks;
  int bh = blockIdx.x / qblocks;
  int h = bh % NHEADS, b = bh / NHEADS;
  int kvh = h / NREP;
  int q0 = qb * 16;

  const bf* Qp = Q  + ((size_t)(b * NHEADS + h) * TLEN + q0) * HD;
  const bf* Kp = Kc + (size_t)(b * NKVH + kvh) * TLEN * HD;
  const bf* Vp = Vt + (size_t)(b * NKVH + kvh) * HD * TLEN;

  // resident Q fragments: 16 x 128 = 4 chunks of K=32
  v16bf qa[4];
#pragma unroll
  for (int c = 0; c < 4; ++c) qa[c] = load_frag_a(Qp + c * 32, HD);

  v8f o[8];
#pragma unroll
  for (int i = 0; i < 8; ++i) o[i] = zero8();
  float rm[8], rl[8];
#pragma unroll
  for (int j = 0; j < 8; ++j) { rm[j] = -3.0e38f; rl[j] = 0.f; }

  const float scale = 0.08838834764831845f;  // 1/sqrt(128)
  int ktiles = (q0 + 16 + 31) / 32;          // causal: keys 0 .. q0+15

  for (int kt = 0; kt < ktiles; ++kt) {
    int kt0 = kt * 32;
    // S = Q * K^T  (K rows are [key][hd] -> direct B-fragment source, ld=HD)
    v8f s0 = zero8(), s1 = zero8();
#pragma unroll
    for (int c = 0; c < 4; ++c) {
      s0 = wmma_bf16(qa[c], load_frag_b(Kp + (size_t)kt0 * HD + c * 32, HD), s0);
      s1 = wmma_bf16(qa[c], load_frag_b(Kp + (size_t)(kt0 + 16) * HD + c * 32, HD), s1);
    }
    // online softmax per row (rows j+half*8 live across the 16 lanes of this half)
#pragma unroll
    for (int j = 0; j < 8; ++j) {
      int qrow = q0 + j + half * 8;
      float v0 = s0[j] * scale; if (kt0 + n      > qrow) v0 = -1.0e9f;
      float v1 = s1[j] * scale; if (kt0 + 16 + n > qrow) v1 = -1.0e9f;
      float t = fmaxf(v0, v1);
#pragma unroll
      for (int off = 8; off; off >>= 1) t = fmaxf(t, __shfl_xor(t, off, 32));
      float mnew = fmaxf(rm[j], t);
      float corr = __expf(rm[j] - mnew);
      float p0 = __expf(v0 - mnew);
      float p1 = __expf(v1 - mnew);
      float ps = p0 + p1;
#pragma unroll
      for (int off = 8; off; off >>= 1) ps += __shfl_xor(ps, off, 32);
      rl[j] = rl[j] * corr + ps;
      rm[j] = mnew;
#pragma unroll
      for (int nt = 0; nt < 8; ++nt) o[nt][j] *= corr;
      // stage P (16x32, row-major) in LDS to reshape C-fragment -> A-fragment
      Pbuf[(j + half * 8) * 32 + n]      = (bf)p0;
      Pbuf[(j + half * 8) * 32 + 16 + n] = (bf)p1;
    }
    asm volatile("s_wait_dscnt 0" ::: "memory");
    FragU pf;
    {
      const bf* prow = &Pbuf[(lane & 15) * 32 + half * 8];
      pf.q[0] = *(const uint4*)(prow);
      pf.q[1] = *(const uint4*)(prow + 16);
    }
    // O += P * V   (Vt rows are [hd][key] -> direct B-fragment source, ld=T)
#pragma unroll
    for (int nt = 0; nt < 8; ++nt) {
      v16bf vb = load_frag_b(Vp + (size_t)(nt * 16) * TLEN + kt0, TLEN);
      o[nt] = wmma_bf16(pf.v, vb, o[nt]);
    }
  }

  // finalize: divide by row sum, write token-major [B*T, DIM]
  bf* Op = Out + ((size_t)(b * TLEN + q0)) * DIM + h * HD;
#pragma unroll
  for (int j = 0; j < 8; ++j) {
    float inv = 1.0f / rl[j];
    int m = j + half * 8;
#pragma unroll
    for (int nt = 0; nt < 8; ++nt) {
      Op[(size_t)m * DIM + nt * 16 + n] = (bf)(o[nt][j] * inv);
    }
  }
}

// ---------------- host launcher ----------------
extern "C" void kernel_launch(void* const* d_in, const int* in_sizes, int n_in,
                              void* d_out, int out_size, void* d_ws, size_t ws_size,
                              hipStream_t stream) {
  const float* x     = (const float*)d_in[0];
  const float* freqs = (const float*)d_in[1];
  /* d_in[2] = mask: causal mask applied analytically in-kernel */
  const float* wq    = (const float*)d_in[3];
  const float* wk    = (const float*)d_in[4];
  const float* wv    = (const float*)d_in[5];
  const float* wo    = (const float*)d_in[6];
  /* d_in[7] = start_pos == 0 */
  float* out = (float*)d_out;

  char* p = (char*)d_ws;
  auto alloc = [&](size_t bytes) -> char* {
    char* r = p;
    p += (bytes + 255) & ~(size_t)255;
    return r;
  };
  bf* xb   = (bf*)alloc((size_t)MTOK * DIM * 2);            // x in bf16
  bf* wqT  = (bf*)alloc((size_t)DIM * DIM * 2);             // [N=4096][K=4096]
  bf* wkT  = (bf*)alloc((size_t)(NKVH * HD) * DIM * 2);     // [1024][4096]
  bf* wvT  = (bf*)alloc((size_t)(NKVH * HD) * DIM * 2);
  bf* woT  = (bf*)alloc((size_t)DIM * DIM * 2);
  bf* qlin = (bf*)alloc((size_t)MTOK * DIM * 2);
  bf* klin = (bf*)alloc((size_t)MTOK * (NKVH * HD) * 2);
  bf* vlin = (bf*)alloc((size_t)MTOK * (NKVH * HD) * 2);
  bf* qr   = (bf*)alloc((size_t)BB * NHEADS * TLEN * HD * 2);
  bf* kr   = (bf*)alloc((size_t)BB * NKVH * TLEN * HD * 2);
  bf* vtb  = (bf*)alloc((size_t)BB * NKVH * HD * TLEN * 2);
  bf* aout = (bf*)alloc((size_t)MTOK * DIM * 2);

  auto blk1d = [](size_t total) { return (unsigned)((total + 255) / 256); };

  // 1) conversions
  size_t nx = (size_t)MTOK * DIM;
  k_f32_to_bf16<<<blk1d(nx), 256, 0, stream>>>(x, xb, nx);
  k_transpose_to_bf16<<<blk1d((size_t)DIM * DIM), 256, 0, stream>>>(wq, wqT, DIM, DIM);
  k_transpose_to_bf16<<<blk1d((size_t)DIM * NKVH * HD), 256, 0, stream>>>(wk, wkT, DIM, NKVH * HD);
  k_transpose_to_bf16<<<blk1d((size_t)DIM * NKVH * HD), 256, 0, stream>>>(wv, wvT, DIM, NKVH * HD);
  k_transpose_to_bf16<<<blk1d((size_t)DIM * DIM), 256, 0, stream>>>(wo, woT, DIM, DIM);

  // 2) QKV projections (WMMA + async-LDS double buffering, 128x128 WG tiles)
  {
    dim3 gq(DIM / 128, MTOK / 128);
    k_gemm_wmma<false><<<gq, 256, 0, stream>>>(xb, wqT, qlin, MTOK, DIM, DIM);
    dim3 gkv((NKVH * HD) / 128, MTOK / 128);
    k_gemm_wmma<false><<<gkv, 256, 0, stream>>>(xb, wkT, klin, MTOK, NKVH * HD, DIM);
    k_gemm_wmma<false><<<gkv, 256, 0, stream>>>(xb, wvT, vlin, MTOK, NKVH * HD, DIM);
  }

  // 3) RoPE + layout
  k_rope_reshape<<<blk1d((size_t)BB * TLEN * NHEADS * HALF_HD), 256, 0, stream>>>(qlin, freqs, qr, NHEADS);
  k_rope_reshape<<<blk1d((size_t)BB * TLEN * NKVH * HALF_HD), 256, 0, stream>>>(klin, freqs, kr, NKVH);
  k_v_transpose<<<blk1d((size_t)BB * TLEN * NKVH * HD), 256, 0, stream>>>(vlin, vtb);

  // 4) flash attention (WMMA), one wave per 16-query block
  {
    unsigned nblocks = (unsigned)(BB * NHEADS * (TLEN / 16));
    k_flash_attn<<<nblocks, 32, 0, stream>>>(qr, kr, vtb, aout);
  }

  // 5) output projection (WMMA, f32 out)
  {
    dim3 go(DIM / 128, MTOK / 128);
    k_gemm_wmma<true><<<go, 256, 0, stream>>>(aout, woT, out, MTOK, DIM, DIM);
  }
}
